// GATNodeFeatures_88527865905191
// MI455X (gfx1250) — matrix operand order repaired and checked by reading, not compiled, not run on hardware
//
#include <hip/hip_runtime.h>

// ---------- types ----------
typedef _Float16 v16h __attribute__((ext_vector_type(16)));
typedef float    v8f  __attribute__((ext_vector_type(8)));
typedef float    f4   __attribute__((ext_vector_type(4)));

#define HIDDEN   128
#define HEADS    8
#define HD       16
#define BATCH    2
#define V        768
#define MROWS    (BATCH * V)      // 1536
#define SPITCH   776              // 768 + 8 pad (breaks 64-bank alignment)

__device__ __forceinline__ v8f wmma_f16(v16h a, v16h b, v8f c) {
  // (neg_a, A, neg_b, B, c_mod, C, reuse_a, reuse_b)
  return __builtin_amdgcn_wmma_f32_16x16x32_f16(false, a, false, b, (short)0, c,
                                                false, false);
}

// A fragment (16x32, f16) from row-major f32 source, leading dim ld.
// 16-bit A layout: lanes 0-15 / 16-31 both carry M = lane&15;
// VGPR j: k = (j>>2)*16 + half*8 + (j&3)*2 (+0/+1 in lo/hi half-dword).
__device__ __forceinline__ v16h load_a_f32(const float* __restrict__ src,
                                           int ld, int row0, int k0) {
  int lane  = threadIdx.x & 31;
  int m     = row0 + (lane & 15);
  int halfl = (lane >> 4) & 1;
  v16h a;
#pragma unroll
  for (int j = 0; j < 8; ++j) {
    int k = k0 + ((j >> 2) << 4) + (halfl << 3) + ((j & 3) << 1);
    a[2 * j]     = (_Float16)src[m * ld + k];
    a[2 * j + 1] = (_Float16)src[m * ld + k + 1];
  }
  return a;
}

// B fragment (32x16, f16) for y = x @ W^T : B[k][n] = W[n][k] (W row-major, ld).
// 16-bit B layout: N = lane&15; VGPR j: k = (j>>2)*16 + half*8 + (j&3)*2 (+0/+1).
__device__ __forceinline__ v16h load_b_wT(const float* __restrict__ W,
                                          int ld, int n0, int k0) {
  int lane  = threadIdx.x & 31;
  int n     = n0 + (lane & 15);
  int halfl = (lane >> 4) & 1;
  v16h b;
#pragma unroll
  for (int j = 0; j < 8; ++j) {
    int k = k0 + ((j >> 2) << 4) + (halfl << 3) + ((j & 3) << 1);
    b[2 * j]     = (_Float16)W[n * ld + k];
    b[2 * j + 1] = (_Float16)W[n * ld + k + 1];
  }
  return b;
}

// ---------------------------------------------------------------------------
// Kernel 1: edge mask = mean(edge_gate, axis=-1) * (1/sqrt(hd))
// Bandwidth-dominant: 604 MB streamed once. One wave per output element:
// 32 lanes x float4 NT loads = 128 contiguous floats, then shuffle reduce.
// ---------------------------------------------------------------------------
__global__ __launch_bounds__(256) void edge_mean_kernel(
    const float* __restrict__ eg, float* __restrict__ maskS, long long nOut) {
  int lane = threadIdx.x & 31;
  long long wave =
      ((long long)blockIdx.x * blockDim.x + threadIdx.x) >> 5;
  long long nWaves = ((long long)gridDim.x * blockDim.x) >> 5;
  const float scale = (1.0f / 128.0f) * 0.25f;  // mean * rsqrt(hd=16)
  for (long long o = wave; o < nOut; o += nWaves) {
    const f4* p = (const f4*)(eg + o * (long long)HIDDEN);
    f4 vv = __builtin_nontemporal_load(p + lane);
    float s = vv.x + vv.y + vv.z + vv.w;
#pragma unroll
    for (int off = 16; off > 0; off >>= 1) s += __shfl_xor(s, off, 32);
    if (lane == 0) maskS[o] = s * scale;
  }
}

// ---------------------------------------------------------------------------
// Kernel 2: fused Q/K/V projections, y = x @ W^T + b, scattered into
//   Q: (b,h,V,hd) row-major      K: (b,h,V,hd) row-major
//   Vt: (b,h,hd,V)  (transposed so attention B-fragments read contiguously)
// One wave per 16x16 output tile; blockIdx.z selects which projection.
// ---------------------------------------------------------------------------
__global__ __launch_bounds__(32) void qkv_kernel(
    const float* __restrict__ x,
    const float* __restrict__ Wq, const float* __restrict__ bq,
    const float* __restrict__ Wk, const float* __restrict__ bk,
    const float* __restrict__ Wv, const float* __restrict__ bv,
    float* __restrict__ Q, float* __restrict__ K, float* __restrict__ Vt) {
  int m0    = blockIdx.x * 16;          // 0..1520
  int n0    = blockIdx.y * 16;          // 0..112
  int which = blockIdx.z;               // 0=Q 1=K 2=V
  const float* W    = (which == 0) ? Wq : (which == 1) ? Wk : Wv;
  const float* bias = (which == 0) ? bq : (which == 1) ? bk : bv;

  v8f acc = {};
#pragma unroll
  for (int k0 = 0; k0 < HIDDEN; k0 += 32) {
    v16h a = load_a_f32(x, HIDDEN, m0, k0);
    v16h b = load_b_wT(W, HIDDEN, n0, k0);
    acc = wmma_f16(a, b, acc);
  }

  int lane  = threadIdx.x & 31;
  int halfl = (lane >> 4) & 1;
  int n     = n0 + (lane & 15);
  float bb  = bias[n];
  int h = n >> 4, d = n & 15;
  int bB = m0 / V;
  int vbase = m0 - bB * V;
#pragma unroll
  for (int j = 0; j < 8; ++j) {
    int vrow = vbase + j + 8 * halfl;
    float y = acc[j] + bb;
    if (which == 2) {
      Vt[(((bB * HEADS + h) * HD) + d) * V + vrow] = y;
    } else {
      float* dst = (which == 0) ? Q : K;
      dst[(((bB * HEADS + h) * V) + vrow) * HD + d] = y;
    }
  }
}

// ---------------------------------------------------------------------------
// Kernel 3: attention for one (b, h, 16-query tile). 8 waves / 256 threads.
// Phase 1: score tiles via WMMA (hd=16 zero-padded to K=32), * mask -> LDS S.
// Phase 2: row softmax (exp stored back into S, 1/rowsum deferred).
// Phase 3: weights @ V via WMMA, K split across waves, LDS partial reduce.
// ---------------------------------------------------------------------------
__global__ __launch_bounds__(256) void attn_kernel(
    const float* __restrict__ Q, const float* __restrict__ K,
    const float* __restrict__ Vt, const float* __restrict__ maskS,
    float* __restrict__ att) {
  __shared__ float S[16 * SPITCH];      // 49,664 B
  __shared__ float partial[8 * 256];    //  8,192 B
  __shared__ float rowinv[16];

  int q0 = blockIdx.x * 16;             // query tile base
  int h  = blockIdx.y;
  int bB = blockIdx.z;
  int wave  = threadIdx.x >> 5;
  int lane  = threadIdx.x & 31;
  int halfl = (lane >> 4) & 1;
  int lo    = lane & 15;

  const float* Qbh  = Q  + (size_t)(bB * HEADS + h) * V * HD;
  const float* Kbh  = K  + (size_t)(bB * HEADS + h) * V * HD;
  const float* Vbh  = Vt + (size_t)(bB * HEADS + h) * HD * V;
  const float* Mrow = maskS + ((long long)(bB * V + q0)) * V;

  // ---- Phase 1: scores -------------------------------------------------
  // A fragment of Q (k = d, zero-padded for d>=16); invariant across key tiles.
  v16h aq;
#pragma unroll
  for (int j = 0; j < 8; ++j) {
    int k = ((j >> 2) << 4) + (halfl << 3) + ((j & 3) << 1);
    _Float16 e0 = (_Float16)0.f, e1 = (_Float16)0.f;
    if (k < HD) {
      e0 = (_Float16)Qbh[(q0 + lo) * HD + k];
      e1 = (_Float16)Qbh[(q0 + lo) * HD + k + 1];
    }
    aq[2 * j] = e0;
    aq[2 * j + 1] = e1;
  }
  for (int kt = wave; kt < V / 16; kt += 8) {
    int key0 = kt * 16;
    v16h bkf;
#pragma unroll
    for (int j = 0; j < 8; ++j) {
      int k = ((j >> 2) << 4) + (halfl << 3) + ((j & 3) << 1);   // k = d
      _Float16 e0 = (_Float16)0.f, e1 = (_Float16)0.f;
      if (k < HD) {
        e0 = (_Float16)Kbh[(key0 + lo) * HD + k];
        e1 = (_Float16)Kbh[(key0 + lo) * HD + k + 1];
      }
      bkf[2 * j] = e0;
      bkf[2 * j + 1] = e1;
    }
    v8f c = {};
    c = wmma_f16(aq, bkf, c);
#pragma unroll
    for (int j = 0; j < 8; ++j) {
      int mrow = j + 8 * halfl;
      int key  = key0 + lo;
      S[mrow * SPITCH + key] = c[j] * Mrow[(long long)mrow * V + key];
    }
  }
  __syncthreads();

  // ---- Phase 2: softmax (rows wave, wave+8) ----------------------------
  for (int r = wave; r < 16; r += 8) {
    float mx = -1e30f;
    for (int c = lane; c < V; c += 32) mx = fmaxf(mx, S[r * SPITCH + c]);
#pragma unroll
    for (int off = 16; off > 0; off >>= 1)
      mx = fmaxf(mx, __shfl_xor(mx, off, 32));
    float sum = 0.f;
    for (int c = lane; c < V; c += 32) {
      float e = __expf(S[r * SPITCH + c] - mx);
      S[r * SPITCH + c] = e;          // unnormalized weight, f32 in LDS
      sum += e;
    }
#pragma unroll
    for (int off = 16; off > 0; off >>= 1) sum += __shfl_xor(sum, off, 32);
    if (lane == 0) rowinv[r] = 1.0f / sum;
  }
  __syncthreads();

  // ---- Phase 3: attended = softmax(S) @ V ------------------------------
  v8f acc = {};
  for (int kc = wave; kc < V / 32; kc += 8) {   // 24 chunks, 3 per wave
    int key0 = kc * 32;
    v16h a;
#pragma unroll
    for (int j = 0; j < 8; ++j) {
      int k = key0 + ((j >> 2) << 4) + (halfl << 3) + ((j & 3) << 1);
      a[2 * j]     = (_Float16)S[lo * SPITCH + k];
      a[2 * j + 1] = (_Float16)S[lo * SPITCH + k + 1];
    }
    v16h b;
#pragma unroll
    for (int j = 0; j < 8; ++j) {
      int k = key0 + ((j >> 2) << 4) + (halfl << 3) + ((j & 3) << 1);   // key index; n = d = lo
      b[2 * j]     = (_Float16)Vbh[lo * V + k];
      b[2 * j + 1] = (_Float16)Vbh[lo * V + k + 1];
    }
    acc = wmma_f16(a, b, acc);
  }
#pragma unroll
  for (int j = 0; j < 8; ++j) partial[wave * 256 + lane * 8 + j] = acc[j];
  __syncthreads();

  if (wave == 0) {
#pragma unroll
    for (int j = 0; j < 8; ++j) {
      float t = 0.f;
#pragma unroll
      for (int w = 0; w < 8; ++w) t += partial[w * 256 + lane * 8 + j];
      int mrow = j + 8 * halfl;
      t *= rowinv[mrow];
      att[((long long)(bB * V + q0 + mrow)) * HIDDEN + h * HD + lo] = t;
    }
  }
}

// ---------------------------------------------------------------------------
// Kernel 4: out = attended @ Wo^T + bo, (B,V,128) f32.
// ---------------------------------------------------------------------------
__global__ __launch_bounds__(32) void outproj_kernel(
    const float* __restrict__ attn, const float* __restrict__ Wo,
    const float* __restrict__ bo, float* __restrict__ out) {
  int m0 = blockIdx.x * 16;
  int n0 = blockIdx.y * 16;
  v8f acc = {};
#pragma unroll
  for (int k0 = 0; k0 < HIDDEN; k0 += 32) {
    v16h a = load_a_f32(attn, HIDDEN, m0, k0);
    v16h b = load_b_wT(Wo, HIDDEN, n0, k0);
    acc = wmma_f16(a, b, acc);
  }
  int lane  = threadIdx.x & 31;
  int halfl = (lane >> 4) & 1;
  int n     = n0 + (lane & 15);
  float bb  = bo[n];
#pragma unroll
  for (int j = 0; j < 8; ++j) {
    int m = m0 + j + 8 * halfl;
    out[m * HIDDEN + n] = acc[j] + bb;
  }
}

// ---------------------------------------------------------------------------
extern "C" void kernel_launch(void* const* d_in, const int* in_sizes, int n_in,
                              void* d_out, int out_size, void* d_ws,
                              size_t ws_size, hipStream_t stream) {
  (void)in_sizes; (void)n_in; (void)out_size; (void)ws_size;
  const float* x  = (const float*)d_in[0];
  const float* eg = (const float*)d_in[1];
  const float* Wq = (const float*)d_in[2];
  const float* bq = (const float*)d_in[3];
  const float* Wk = (const float*)d_in[4];
  const float* bk = (const float*)d_in[5];
  const float* Wv = (const float*)d_in[6];
  const float* bv = (const float*)d_in[7];
  const float* Wo = (const float*)d_in[8];
  const float* bo = (const float*)d_in[9];
  float* out = (float*)d_out;

  float* ws    = (float*)d_ws;
  const long long nMask = (long long)BATCH * V * V;          // 1,179,648
  const long long nQKV  = (long long)BATCH * HEADS * V * HD; //   196,608
  float* maskS = ws;
  float* Q     = maskS + nMask;
  float* K     = Q + nQKV;
  float* Vt    = K + nQKV;
  float* att   = Vt + nQKV;

  edge_mean_kernel<<<2048, 256, 0, stream>>>(eg, maskS, nMask);
  qkv_kernel<<<dim3(MROWS / 16, HIDDEN / 16, 3), 32, 0, stream>>>(
      x, Wq, bq, Wk, bk, Wv, bv, Q, K, Vt);
  attn_kernel<<<dim3(V / 16, HEADS, BATCH), 256, 0, stream>>>(Q, K, Vt, maskS,
                                                              att);
  outproj_kernel<<<dim3(MROWS / 16, HIDDEN / 16), 32, 0, stream>>>(att, Wo, bo,
                                                                   out);
}
